// BidirectionalMambaBlock_10625749090875
// MI455X (gfx1250) — matrix operand order repaired and checked by reading, not compiled, not run on hardware
//
#include <hip/hip_runtime.h>
#include <hip/hip_bf16.h>

#define B_  4
#define T_  2048
#define DM  256
#define DI  512
#define DS_ 16
#define DR  16
#define DRP 32            // dt K padded to one WMMA K-step
#define MROWS (B_*T_)

typedef __attribute__((ext_vector_type(16))) __bf16 v16bf;
typedef __attribute__((ext_vector_type(8)))  __bf16 v8bf;
typedef __attribute__((ext_vector_type(8)))  float  v8f;

static __device__ __forceinline__ unsigned short f32_to_bf16_bits(float f) {
  unsigned int u = __float_as_uint(f);
  unsigned int r = 0x7FFFu + ((u >> 16) & 1u);   // round-to-nearest-even
  u += r;
  return (unsigned short)(u >> 16);
}

// ---------------- elementwise f32 -> bf16 (weights) ----------------
__global__ void k_f32_to_bf16(const float* __restrict__ src,
                              unsigned short* __restrict__ dst, int n) {
  int i = blockIdx.x * blockDim.x + threadIdx.x;
  if (i < n) dst[i] = f32_to_bf16_bits(src[i]);
}

// ---- f32[N,srcK] -> bf16[N,dstK] with zero pad in K (for dt_w) ----
__global__ void k_f32_to_bf16_padK(const float* __restrict__ src,
                                   unsigned short* __restrict__ dst,
                                   int n, int srcK, int dstK) {
  int i = blockIdx.x * blockDim.x + threadIdx.x;
  if (i >= n) return;
  int r = i / dstK, c = i % dstK;
  dst[i] = (c < srcK) ? f32_to_bf16_bits(src[r * srcK + c]) : (unsigned short)0;
}

// ---------------- LayerNorm over DM=256, bf16 output ----------------
__global__ void k_layernorm_bf16(const float* __restrict__ x,
                                 const float* __restrict__ w,
                                 const float* __restrict__ b,
                                 unsigned short* __restrict__ out) {
  __shared__ float s1[DM], s2[DM];
  const int row = blockIdx.x;
  const int c   = threadIdx.x;
  float v = x[(size_t)row * DM + c];
  s1[c] = v; s2[c] = v * v;
  __syncthreads();
  for (int st = DM / 2; st > 0; st >>= 1) {
    if (c < st) { s1[c] += s1[c + st]; s2[c] += s2[c + st]; }
    __syncthreads();
  }
  float mu  = s1[0] * (1.0f / DM);
  float var = s2[0] * (1.0f / DM) - mu * mu;
  float xn  = (v - mu) * rsqrtf(var + 1e-5f) * w[c] + b[c];
  out[(size_t)row * DM + c] = f32_to_bf16_bits(xn);
}

// ======== register-blocked WMMA GEMM: C = A(bf16) @ W(bf16,[N,K])^T ========
// One wave computes an (MT*16) x (NT*16) block of C: A fragments reused NT
// times, B fragments reused MT times -> 2 b128 loads per wmma.
// K compile-time (multiple of 32), fully unrolled; no predication, EXEC all-1.
template <int K, int MT, int NT>
__global__ void k_wmma_gemm(const unsigned short* __restrict__ A,
                            const unsigned short* __restrict__ W,
                            float* __restrict__ Cf,
                            unsigned short* __restrict__ Cb,
                            const float* __restrict__ bias,
                            const float* __restrict__ resid,
                            int M, int N, int lda, int ldb, int ldc,
                            int flipA, int flipC, int do_softplus) {
  const int lane = threadIdx.x & 31;
  const int wid  = threadIdx.x >> 5;
  int g = blockIdx.x * (blockDim.x >> 5) + wid;
  const int mg = M / (16 * MT);
  const int ng = N / (16 * NT);
  if (g >= mg * ng) g = 0;                 // wave-uniform safety clamp
  const int m0 = (g % mg) * 16 * MT;
  const int n0 = (g / mg) * 16 * NT;

  // A operand: 16x32 bf16; lane -> M = lane%16; per lane two contiguous
  // 8-elt chunks at +0/+16 (element i -> K = k0 + i + 8*(i>=8) + 8*(lane>=16)).
  const unsigned short* Arow[MT];
#pragma unroll
  for (int mt = 0; mt < MT; ++mt) {
    int mA = m0 + mt * 16 + (lane & 15);
    if (flipA) { int bb = mA / T_; int tt = mA % T_; mA = bb * T_ + (T_ - 1 - tt); }
    Arow[mt] = A + (size_t)mA * lda + ((lane >= 16) ? 8 : 0);
  }
  // B operand: 32x16 bf16; lane -> N = lane%16; per lane two contiguous
  // 8-elt chunks at +0/+16 (element i -> K = k0 + i + 8*(i>=8) + 8*(lane>=16)).
  const unsigned short* Brow[NT];
#pragma unroll
  for (int nt = 0; nt < NT; ++nt)
    Brow[nt] = W + (size_t)(n0 + nt * 16 + (lane & 15)) * ldb + ((lane >= 16) ? 8 : 0);

  const v8f vzero = {};
  v8f acc[MT][NT];
#pragma unroll
  for (int mt = 0; mt < MT; ++mt)
#pragma unroll
    for (int nt = 0; nt < NT; ++nt) acc[mt][nt] = vzero;

#pragma unroll
  for (int k0 = 0; k0 < K; k0 += 32) {
    v16bf av[MT], bv[NT];
#pragma unroll
    for (int mt = 0; mt < MT; ++mt) {
      const v8bf* pa = (const v8bf*)(Arow[mt] + k0);
      av[mt] = __builtin_shufflevector(pa[0], pa[2],
                                       0,1,2,3,4,5,6,7,8,9,10,11,12,13,14,15);
    }
#pragma unroll
    for (int nt = 0; nt < NT; ++nt) {
      const v8bf* pb = (const v8bf*)(Brow[nt] + k0);
      bv[nt] = __builtin_shufflevector(pb[0], pb[2],
                                       0,1,2,3,4,5,6,7,8,9,10,11,12,13,14,15);
    }
#pragma unroll
    for (int mt = 0; mt < MT; ++mt)
#pragma unroll
      for (int nt = 0; nt < NT; ++nt)
        acc[mt][nt] = __builtin_amdgcn_wmma_f32_16x16x32_bf16(
            false, av[mt], false, bv[nt], (short)0, acc[mt][nt], false, false);
  }

  // C layout: VGPR j -> M = j + 8*(lane>=16), N = lane%16 within each subtile
#pragma unroll
  for (int nt = 0; nt < NT; ++nt) {
    const int cn = n0 + nt * 16 + (lane & 15);
    const float bvl = bias ? bias[cn] : 0.0f;
#pragma unroll
    for (int mt = 0; mt < MT; ++mt) {
#pragma unroll
      for (int j = 0; j < 8; ++j) {
        int cm = m0 + mt * 16 + j + ((lane >= 16) ? 8 : 0);
        int dr = cm;
        if (flipC) { int bb = cm / T_; int tt = cm % T_; dr = bb * T_ + (T_ - 1 - tt); }
        float v = acc[mt][nt][j] + bvl;
        if (do_softplus) v = (v > 20.0f) ? v : log1pf(expf(v));
        if (resid) v += resid[(size_t)dr * ldc + cn];
        if (Cf) Cf[(size_t)dr * ldc + cn] = v;
        else    Cb[(size_t)dr * ldc + cn] = f32_to_bf16_bits(v);
      }
    }
  }
}

// ---------------- causal depthwise conv4 + SiLU ----------------
__global__ void k_conv_silu(const float* __restrict__ xz,
                            const float* __restrict__ cw,
                            const float* __restrict__ cb,
                            float* __restrict__ u32,
                            unsigned short* __restrict__ ubf) {
  int idx = blockIdx.x * blockDim.x + threadIdx.x;   // [MROWS*DI]
  int d   = idx & (DI - 1);
  int row = idx >> 9;
  int bb = row / T_, tt = row % T_;
  float acc = cb[d];
#pragma unroll
  for (int k = 0; k < 4; ++k) {
    int ts = tt + k - 3;
    float uv = (ts >= 0) ? xz[(size_t)(bb * T_ + ts) * (2 * DI) + d] : 0.0f;
    acc += cw[d * 4 + k] * uv;
  }
  acc = acc / (1.0f + expf(-acc));                   // SiLU
  u32[idx] = acc;
  ubf[idx] = f32_to_bf16_bits(acc);
}

// ------- extract dt columns of x_dbl to bf16, zero-padded K 16->32 -------
__global__ void k_dt_extract(const float* __restrict__ xdbl,
                             unsigned short* __restrict__ dtbf) {
  int idx = blockIdx.x * blockDim.x + threadIdx.x;   // MROWS*DRP
  int r = idx >> 5, c = idx & 31;
  dtbf[idx] = (c < DR) ? f32_to_bf16_bits(xdbl[(size_t)r * 48 + c])
                       : (unsigned short)0;
}

// ------- selective scan: lane = (d, n) state, shfl_xor reduce over n -------
__global__ void k_scan(const float* __restrict__ delta,
                       const float* __restrict__ u32,
                       const float* __restrict__ xdbl,
                       const float* __restrict__ xz,     // z gate at cols [DI, 2*DI)
                       const float* __restrict__ A_log,
                       const float* __restrict__ Dskip,
                       unsigned short* __restrict__ ybf) {
  // grid = B_*32 blocks, 256 threads = 16 channels x 16 states
  const int lane_n = threadIdx.x & 15;
  const int dloc   = threadIdx.x >> 4;
  const int d  = (blockIdx.x & 31) * 16 + dloc;
  const int bb = blockIdx.x >> 5;
  const float A  = -expf(A_log[d * DS_ + lane_n]);
  const float dsk = Dskip[d];
  float h = 0.0f;
  const size_t base = (size_t)bb * T_;
  for (int t = 0; t < T_; ++t) {
    const size_t row = base + t;
    const float dv = delta[row * DI + d];
    const float uv = u32[row * DI + d];
    const float Bn = xdbl[row * 48 + 16 + lane_n];
    const float Cn = xdbl[row * 48 + 32 + lane_n];
    h = expf(dv * A) * h + dv * Bn * uv;
    float y = h * Cn;
    y += __shfl_xor(y, 1, 32);
    y += __shfl_xor(y, 2, 32);
    y += __shfl_xor(y, 4, 32);
    y += __shfl_xor(y, 8, 32);
    if (lane_n == 0) {
      y += uv * dsk;
      float zv = xz[row * (2 * DI) + DI + d];
      y *= zv / (1.0f + expf(-zv));                  // * SiLU(z)
      ybf[row * DI + d] = f32_to_bf16_bits(y);
    }
  }
}

extern "C" void kernel_launch(void* const* d_in, const int* in_sizes, int n_in,
                              void* d_out, int out_size, void* d_ws, size_t ws_size,
                              hipStream_t stream) {
  (void)in_sizes; (void)n_in; (void)out_size; (void)ws_size;
  const float* x      = (const float*)d_in[0];
  const float* norm_w = (const float*)d_in[1];
  const float* norm_b = (const float*)d_in[2];
  const float* fus_w  = (const float*)d_in[3];
  const float* fus_b  = (const float*)d_in[4];

  char* ws = (char*)d_ws;
  size_t off = 0;
  auto alloc = [&](size_t bytes) -> char* {
    char* p = ws + off;
    off = (off + bytes + 255) & ~(size_t)255;
    return p;
  };

  unsigned short* xn_bf = (unsigned short*)alloc((size_t)MROWS * DM * 2);
  unsigned short* w_in  = (unsigned short*)alloc((size_t)2 * (2 * DI) * DM * 2);
  unsigned short* w_xp  = (unsigned short*)alloc((size_t)2 * 48 * DI * 2);
  unsigned short* w_dt  = (unsigned short*)alloc((size_t)2 * DI * DRP * 2);
  unsigned short* w_out = (unsigned short*)alloc((size_t)2 * DM * DI * 2);
  unsigned short* w_fus = (unsigned short*)alloc((size_t)DM * (2 * DM) * 2);
  float*          xz    = (float*)alloc((size_t)MROWS * (2 * DI) * 4);
  float*          u32   = (float*)alloc((size_t)MROWS * DI * 4);
  unsigned short* ubf   = (unsigned short*)alloc((size_t)MROWS * DI * 2);
  float*          xdbl  = (float*)alloc((size_t)MROWS * 48 * 4);
  unsigned short* dtbf  = (unsigned short*)alloc((size_t)MROWS * DRP * 2);
  float*          delta = (float*)alloc((size_t)MROWS * DI * 4);
  unsigned short* ybf   = (unsigned short*)alloc((size_t)MROWS * DI * 2);
  unsigned short* xcat  = (unsigned short*)alloc((size_t)MROWS * (2 * DM) * 2);

  auto cvt = [&](const float* s, unsigned short* dst, int n) {
    k_f32_to_bf16<<<(n + 255) / 256, 256, 0, stream>>>(s, dst, n);
  };
  cvt(fus_w, w_fus, DM * 2 * DM);
  for (int dir = 0; dir < 2; ++dir) {
    const int base = 5 + dir * 9;
    cvt((const float*)d_in[base + 0], w_in  + (size_t)dir * (2 * DI) * DM, 2 * DI * DM);
    cvt((const float*)d_in[base + 3], w_xp  + (size_t)dir * 48 * DI,        48 * DI);
    cvt((const float*)d_in[base + 8], w_out + (size_t)dir * DM * DI,        DM * DI);
    // dt_w [512,16] -> bf16 [512,32] zero-padded
    k_f32_to_bf16_padK<<<(DI * DRP + 255) / 256, 256, 0, stream>>>(
        (const float*)d_in[base + 4], w_dt + (size_t)dir * DI * DRP,
        DI * DRP, DR, DRP);
  }

  k_layernorm_bf16<<<MROWS, DM, 0, stream>>>(x, norm_w, norm_b, xn_bf);

  for (int dir = 0; dir < 2; ++dir) {
    const int base = 5 + dir * 9;
    const float* conv_w = (const float*)d_in[base + 1];
    const float* conv_b = (const float*)d_in[base + 2];
    const float* dt_b   = (const float*)d_in[base + 5];
    const float* A_log  = (const float*)d_in[base + 6];
    const float* Dskip  = (const float*)d_in[base + 7];

    // xz = (flip?)xn @ in_w^T            [8192 x 1024], K=256, 2x2 blocking
    {
      int waves = (MROWS / 32) * ((2 * DI) / 32);
      k_wmma_gemm<DM, 2, 2><<<waves / 4, 128, 0, stream>>>(
          xn_bf, w_in + (size_t)dir * (2 * DI) * DM, xz, nullptr, nullptr, nullptr,
          MROWS, 2 * DI, DM, DM, 2 * DI, dir, 0, 0);
    }
    // u = SiLU(conv4(xz[:, :512]))       [8192 x 512]
    k_conv_silu<<<(MROWS * DI) / 256, 256, 0, stream>>>(xz, conv_w, conv_b, u32, ubf);
    // x_dbl = u @ xproj^T                [8192 x 48], K=512, 2x3 blocking
    {
      int waves = (MROWS / 32) * (48 / 48);
      k_wmma_gemm<DI, 2, 3><<<waves / 4, 128, 0, stream>>>(
          ubf, w_xp + (size_t)dir * 48 * DI, xdbl, nullptr, nullptr, nullptr,
          MROWS, 48, DI, DI, 48, 0, 0, 0);
    }
    k_dt_extract<<<(MROWS * DRP) / 256, 256, 0, stream>>>(xdbl, dtbf);
    // delta = softplus(dt @ dt_w^T + dt_b) [8192 x 512], K=32 (padded), 2x2
    {
      int waves = (MROWS / 32) * (DI / 32);
      k_wmma_gemm<DRP, 2, 2><<<waves / 4, 128, 0, stream>>>(
          dtbf, w_dt + (size_t)dir * DI * DRP, delta, nullptr, dt_b, nullptr,
          MROWS, DI, DRP, DRP, DI, 0, 0, 1);
    }
    // selective scan (+D skip, *SiLU(z)) -> ybf
    k_scan<<<B_ * 32, 256, 0, stream>>>(delta, u32, xdbl, xz, A_log, Dskip, ybf);
    // out-proj -> x_cat columns (bwd: unflip rows) [8192 x 256], K=512, 2x2
    {
      int waves = (MROWS / 32) * (DM / 32);
      k_wmma_gemm<DI, 2, 2><<<waves / 4, 128, 0, stream>>>(
          ybf, w_out + (size_t)dir * DM * DI, nullptr, xcat + dir * DM, nullptr, nullptr,
          MROWS, DM, DI, DI, 2 * DM, 0, dir, 0);
    }
  }

  // out = x_cat @ fus_w^T + fus_b + residual(x) [8192 x 256], K=512, 2x2
  {
    int waves = (MROWS / 32) * (DM / 32);
    k_wmma_gemm<2 * DM, 2, 2><<<waves / 4, 128, 0, stream>>>(
        xcat, w_fus, (float*)d_out, nullptr, fus_b, x,
        MROWS, DM, 2 * DM, 2 * DM, DM, 0, 0, 0);
  }
}
